// CTM_20486994002684
// MI455X (gfx1250) — compile-verified, matrix operand order checked by reading, not run on hardware
//
#include <hip/hip_runtime.h>
#include <hip/hip_bf16.h>
#include <stdint.h>

// ---------------------------------------------------------------------------
// Problem constants (from reference setup_inputs)
// ---------------------------------------------------------------------------
constexpr int B   = 4;
constexpr int N   = 3136;     // tokens (= 56*56)
constexpr int C   = 64;       // input channels
constexpr int C2  = 128;      // conv / feature channels
constexpr int H   = 56, W = 56;
constexpr int H2  = 28, W2 = 28;
constexpr int P2  = H2 * W2;  // 784 output positions
constexpr int NS  = 784;      // ceil(N * 0.25)
constexpr int KNN = 5;
constexpr int NT  = N / 16;   // 196 row tiles
constexpr int CT  = NS / 16;  // 49 center tiles

typedef __bf16 bf16_t;
typedef __attribute__((ext_vector_type(16))) __bf16 v16bf;
typedef __attribute__((ext_vector_type(8)))  __bf16 v8bf;
typedef __attribute__((ext_vector_type(8)))  float  v8f;
typedef __attribute__((ext_vector_type(4)))  int    v4i;

#define FINF 3.0e38f

// ---------------------------------------------------------------------------
// WMMA fragment loader: 16-bit A/B fragment for v_wmma_f32_16x16x32_bf16.
// Per ISA 7.12.2: lane L holds matrix row (A) / col (B) = L&15, with
// kbase = (L>>4)*8; elements 0..7 cover K=kbase..kbase+7 (contiguous),
// elements 8..15 cover K=kbase+16..kbase+23 (contiguous). Two b128 loads
// (global_load_b128 or ds_load_b128 depending on the pointer).
// ---------------------------------------------------------------------------
__device__ inline v16bf load_frag(const bf16_t* row, int off) {
    v16bf f;
    v8bf* p = reinterpret_cast<v8bf*>(&f);
    p[0] = *reinterpret_cast<const v8bf*>(row + off);
    p[1] = *reinterpret_cast<const v8bf*>(row + off + 16);
    return f;
}

__device__ inline v8f wmma_bf16(v16bf a, v16bf b, v8f c) {
    return __builtin_amdgcn_wmma_f32_16x16x32_bf16(false, a, false, b,
                                                   (short)0, c, false, false);
}

// ---------------------------------------------------------------------------
// CDNA5 async global->LDS staging (ASYNCcnt path), with sync fallback.
// Probe result: builtin exists; first param is 'v4i *' (16B vector pointee).
// ---------------------------------------------------------------------------
#if defined(__has_builtin)
#if __has_builtin(__builtin_amdgcn_global_load_async_to_lds_b128)
#define HAVE_ASYNC_LDS 1
#endif
#endif
#ifndef HAVE_ASYNC_LDS
#define HAVE_ASYNC_LDS 0
#endif

typedef __attribute__((address_space(1))) v4i as1_v4i;
typedef __attribute__((address_space(3))) v4i as3_v4i;

__device__ inline void async_copy16(const bf16_t* g, bf16_t* l) {
#if HAVE_ASYNC_LDS
    __builtin_amdgcn_global_load_async_to_lds_b128(
        (as1_v4i*)g, (as3_v4i*)l, 0, 0);
#else
    *reinterpret_cast<v8bf*>(l) = *reinterpret_cast<const v8bf*>(g);
#endif
}

__device__ inline void async_wait_all() {
#if HAVE_ASYNC_LDS
#if defined(__has_builtin) && __has_builtin(__builtin_amdgcn_s_wait_asynccnt)
    __builtin_amdgcn_s_wait_asynccnt(0);
#else
    asm volatile("s_wait_asynccnt 0" ::: "memory");
#endif
#endif
}

// stage one contiguous 16-row x 128-col bf16 panel (4 KB) into LDS:
// 8 x b128 per lane-group (32 lanes x 16 B = 512 B per instruction)
__device__ inline void stage_panel(const bf16_t* gpanel, bf16_t* lpanel, int lane) {
#pragma unroll
    for (int chunk = 0; chunk < 8; ++chunk) {
        int off = chunk * 256 + lane * 8;    // bf16 elements; 16 B aligned
        async_copy16(gpanel + off, lpanel + off);
    }
}

// round-half-to-even grid index, matches jnp.round
__device__ inline int grid_idx(float lx, float ly, int Hh, int Ww) {
    lx = fminf(fmaxf(lx, -1.f), 1.f) * 0.5f + 0.5f;
    ly = fminf(fmaxf(ly, -1.f), 1.f) * 0.5f + 0.5f;
    int ix = __float2int_rn(lx * (float)(Ww - 1));
    int iy = __float2int_rn(ly * (float)(Hh - 1));
    ix = min(max(ix, 0), Ww - 1);
    iy = min(max(iy, 0), Hh - 1);
    return iy * Ww + ix;
}

// ---------------------------------------------------------------------------
// generic fill
// ---------------------------------------------------------------------------
__global__ void k_fill(float* p, float v, int n) {
    int i = blockIdx.x * blockDim.x + threadIdx.x;
    int stride = gridDim.x * blockDim.x;
    for (; i < n; i += stride) p[i] = v;
}

// convert fp32 -> bf16
__global__ void k_cvt(const float* __restrict__ src, bf16_t* __restrict__ dst, int n) {
    int i = blockIdx.x * blockDim.x + threadIdx.x;
    int stride = gridDim.x * blockDim.x;
    for (; i < n; i += stride) dst[i] = (bf16_t)src[i];
}

// ---------------------------------------------------------------------------
// token2map: cell counts, then normalized scatter of x into 56x56 grid (CL)
// ---------------------------------------------------------------------------
__global__ void k_t2m_w(const float* __restrict__ loc, float* allw) {
    int i = blockIdx.x * blockDim.x + threadIdx.x;
    if (i >= B * N) return;
    int b = i / N;
    int cell = grid_idx(loc[i * 2], loc[i * 2 + 1], H, W);
    atomicAdd(&allw[b * (H * W) + cell], 1.0f);
}

__global__ void k_t2m_s(const float* __restrict__ x, const float* __restrict__ loc,
                        const int* __restrict__ idxagg, const float* __restrict__ allw,
                        float* xmap) {
    int i = blockIdx.x;            // point 0..B*N-1
    int c = threadIdx.x;           // 0..63
    int b = i / N;
    int cell = grid_idx(loc[i * 2], loc[i * 2 + 1], H, W);
    int tok  = idxagg[i];
    float val = 1.0f / (allw[b * (H * W) + cell] + 1e-6f);
    atomicAdd(&xmap[((size_t)(b * (H * W) + cell)) * C + c],
              x[((size_t)(b * N + tok)) * C + c] * val);
}

// ---------------------------------------------------------------------------
// 3x3 stride-2 conv, pad 1, 64->128, channels-last in/out, LDS-staged taps
// ---------------------------------------------------------------------------
__global__ void k_conv(const float* __restrict__ xmap, const float* __restrict__ w,
                       const float* __restrict__ bias, float* __restrict__ out) {
    int p = blockIdx.x % P2;
    int b = blockIdx.x / P2;
    int oy = p / W2, ox = p % W2;
    __shared__ float s[9 * C];
    int t = threadIdx.x;           // 0..127
    for (int idx = t; idx < 9 * C; idx += C2) {
        int tap = idx >> 6;        // 0..8 (C == 64)
        int c   = idx & (C - 1);
        int ky = tap / 3, kx = tap % 3;
        int iy = oy * 2 - 1 + ky, ix = ox * 2 - 1 + kx;
        float v = 0.f;
        if (iy >= 0 && iy < H && ix >= 0 && ix < W)
            v = xmap[((size_t)((b * H + iy) * W + ix)) * C + c];
        s[idx] = v;
    }
    __syncthreads();
    int o = t;
    float acc = bias[o];
    const float* wrow = w + (size_t)o * (C * 9);   // [c][ky][kx]
    for (int c = 0; c < C; ++c) {
        float xv0 = s[0 * C + c], xv1 = s[1 * C + c], xv2 = s[2 * C + c];
        float xv3 = s[3 * C + c], xv4 = s[4 * C + c], xv5 = s[5 * C + c];
        float xv6 = s[6 * C + c], xv7 = s[7 * C + c], xv8 = s[8 * C + c];
        const float* wc = wrow + c * 9;
        acc += xv0 * wc[0] + xv1 * wc[1] + xv2 * wc[2]
             + xv3 * wc[3] + xv4 * wc[4] + xv5 * wc[5]
             + xv6 * wc[6] + xv7 * wc[7] + xv8 * wc[8];
    }
    out[((size_t)(b * P2 + p)) * C2 + o] = acc;
}

// ---------------------------------------------------------------------------
// map2token: per-token weight sums, then normalized gather-scatter into xt
// ---------------------------------------------------------------------------
__global__ void k_m2t_w(const int* __restrict__ idxagg, const float* __restrict__ aggw,
                        float* allw) {
    int i = blockIdx.x * blockDim.x + threadIdx.x;
    if (i >= B * N) return;
    int b = i / N;
    atomicAdd(&allw[b * N + idxagg[i]], aggw[i]);
}

__global__ void k_m2t_s(const float* __restrict__ conv_out, const float* __restrict__ loc,
                        const int* __restrict__ idxagg, const float* __restrict__ aggw,
                        const float* __restrict__ allw, float* xt) {
    int i = blockIdx.x;            // point
    int c = threadIdx.x;           // 0..127
    int b = i / N;
    int tok  = idxagg[i];
    int cell = grid_idx(loc[i * 2], loc[i * 2 + 1], H2, W2);
    float v = aggw[i] / (allw[b * N + tok] + 1e-6f);
    atomicAdd(&xt[((size_t)(b * N + tok)) * C2 + c],
              conv_out[((size_t)(b * P2 + cell)) * C2 + c] * v);
}

// ---------------------------------------------------------------------------
// skip projection: xt += x @ skip_w.T  via WMMA bf16 (M=B*N, K=64, N=128)
// ---------------------------------------------------------------------------
__global__ void k_skip(const bf16_t* __restrict__ xb, const bf16_t* __restrict__ wb,
                       float* xt) {
    int it = blockIdx.x;           // 0..783  (M tiles)
    int jt = blockIdx.y;           // 0..7    (N tiles)
    int lane = threadIdx.x;
    int half = lane >> 4, mloc = lane & 15, kbase = half * 8;
    const bf16_t* arow = xb + (size_t)(it * 16 + mloc) * C;
    const bf16_t* brow = wb + (size_t)(jt * 16 + mloc) * C;
    v8f c = {};
#pragma unroll
    for (int kk = 0; kk < 2; ++kk) {
        v16bf a  = load_frag(arow, kk * 32 + kbase);
        v16bf bb = load_frag(brow, kk * 32 + kbase);
        c = wmma_bf16(a, bb, c);
    }
    int col = jt * 16 + mloc;
#pragma unroll
    for (int r = 0; r < 8; ++r) {
        size_t o = (size_t)(it * 16 + half * 8 + r) * C2 + col;
        xt[o] += c[r];
    }
}

// ---------------------------------------------------------------------------
// LayerNorm + conf/weight + bf16 copy + squared norms
// ---------------------------------------------------------------------------
__global__ void k_ln(float* xt, bf16_t* xtb, const float* __restrict__ g,
                     const float* __restrict__ be, const float* __restrict__ cw,
                     const float* __restrict__ cb, float* weight, float* x2) {
    int i = blockIdx.x;            // token
    int c = threadIdx.x;           // 0..127
    __shared__ float red[C2];
    float v = xt[(size_t)i * C2 + c];
    red[c] = v; __syncthreads();
    for (int off = C2 / 2; off; off >>= 1) { if (c < off) red[c] += red[c + off]; __syncthreads(); }
    float mu = red[0] * (1.0f / C2); __syncthreads();
    float d = v - mu;
    red[c] = d * d; __syncthreads();
    for (int off = C2 / 2; off; off >>= 1) { if (c < off) red[c] += red[c + off]; __syncthreads(); }
    float var = red[0] * (1.0f / C2); __syncthreads();
    float xn = d * rsqrtf(var + 1e-5f) * g[c] + be[c];
    xt[(size_t)i * C2 + c] = xn;
    xtb[(size_t)i * C2 + c] = (bf16_t)xn;
    red[c] = xn * cw[c]; __syncthreads();
    for (int off = C2 / 2; off; off >>= 1) { if (c < off) red[c] += red[c + off]; __syncthreads(); }
    float conf = red[0] + cb[0]; __syncthreads();
    red[c] = xn * xn; __syncthreads();
    for (int off = C2 / 2; off; off >>= 1) { if (c < off) red[c] += red[c + off]; __syncthreads(); }
    if (c == 0) { weight[i] = expf(conf); x2[i] = red[0]; }
}

// ---------------------------------------------------------------------------
// DPC-KNN density: per row, 5 smallest squared dists over all N.
// WMMA Gram pass with double-buffered async global->LDS B panels.
// ---------------------------------------------------------------------------
__global__ void k_dens(const bf16_t* __restrict__ xtb, const float* __restrict__ x2,
                       float* density) {
    int b = blockIdx.y, it = blockIdx.x, lane = threadIdx.x;
    int half = lane >> 4, mloc = lane & 15, kbase = half * 8;
    const bf16_t* base = xtb + (size_t)b * N * C2;
    const float*  x2b  = x2 + b * N;
    const bf16_t* arow = base + (size_t)(it * 16 + mloc) * C2;
    v16bf a[4];
#pragma unroll
    for (int kk = 0; kk < 4; ++kk) a[kk] = load_frag(arow, kk * 32 + kbase);

    __shared__ bf16_t panel[2][16 * C2];    // 2 x 4 KB double buffer
    __shared__ float tile[16][16];
    __shared__ float sx2[16];
    float tp[KNN] = {FINF, FINF, FINF, FINF, FINF};
    float x2i = (lane < 16) ? x2b[it * 16 + lane] : 0.f;

    stage_panel(base, panel[0], lane);      // prologue: panel for jt=0

    for (int jt = 0; jt < NT; ++jt) {
        int cur = jt & 1;
        async_wait_all();                   // panel[cur] resident in LDS
        __syncthreads();
        if (jt + 1 < NT)                    // overlap next DMA with compute
            stage_panel(base + (size_t)(jt + 1) * 16 * C2, panel[1 - cur], lane);

        const bf16_t* brow = &panel[cur][mloc * C2];
        v8f c = {};
#pragma unroll
        for (int kk = 0; kk < 4; ++kk) {
            v16bf bb = load_frag(brow, kk * 32 + kbase);   // ds_load_b128 x2
            c = wmma_bf16(a[kk], bb, c);
        }
#pragma unroll
        for (int r = 0; r < 8; ++r) tile[half * 8 + r][mloc] = c[r];
        if (lane < 16) sx2[lane] = x2b[jt * 16 + lane];
        __syncthreads();
        if (lane < 16) {
            for (int col = 0; col < 16; ++col) {
                float d2 = fmaxf(x2i + sx2[col] - 2.f * tile[lane][col], 0.f) * (1.f / C2);
                if (d2 < tp[KNN - 1]) {
                    tp[KNN - 1] = d2;
                    for (int q = KNN - 1; q > 0; --q)
                        if (tp[q] < tp[q - 1]) { float t = tp[q]; tp[q] = tp[q - 1]; tp[q - 1] = t; }
                }
            }
        }
        __syncthreads();
    }
    if (lane < 16) {
        int row = it * 16 + lane;
        float m = (tp[0] + tp[1] + tp[2] + tp[3] + tp[4]) * 0.2f;
        unsigned hsh = (unsigned)(b * N + row) * 2654435761u;
        float noise = (float)(hsh >> 8) * (1e-6f / 16777216.0f);
        density[b * N + row] = expf(-m) + noise;
    }
}

// ---------------------------------------------------------------------------
// masked min distance to denser points + global dist max (second async pass)
// ---------------------------------------------------------------------------
__global__ void k_dind(const bf16_t* __restrict__ xtb, const float* __restrict__ x2,
                       const float* __restrict__ density, float* minMasked,
                       float* distmax) {
    int b = blockIdx.y, it = blockIdx.x, lane = threadIdx.x;
    int half = lane >> 4, mloc = lane & 15, kbase = half * 8;
    const bf16_t* base = xtb + (size_t)b * N * C2;
    const float*  x2b  = x2 + b * N;
    const float*  dnb  = density + b * N;
    const bf16_t* arow = base + (size_t)(it * 16 + mloc) * C2;
    v16bf a[4];
#pragma unroll
    for (int kk = 0; kk < 4; ++kk) a[kk] = load_frag(arow, kk * 32 + kbase);

    __shared__ bf16_t panel[2][16 * C2];
    __shared__ float tile[16][16];
    __shared__ float sx2[16];
    __shared__ float sdn[16];
    __shared__ float smax[16];
    float x2i = (lane < 16) ? x2b[it * 16 + lane] : 0.f;
    float di  = (lane < 16) ? dnb[it * 16 + lane] : 0.f;
    float mn = FINF, mx = 0.f;

    stage_panel(base, panel[0], lane);

    for (int jt = 0; jt < NT; ++jt) {
        int cur = jt & 1;
        async_wait_all();
        __syncthreads();
        if (jt + 1 < NT)
            stage_panel(base + (size_t)(jt + 1) * 16 * C2, panel[1 - cur], lane);

        const bf16_t* brow = &panel[cur][mloc * C2];
        v8f c = {};
#pragma unroll
        for (int kk = 0; kk < 4; ++kk) {
            v16bf bb = load_frag(brow, kk * 32 + kbase);
            c = wmma_bf16(a[kk], bb, c);
        }
#pragma unroll
        for (int r = 0; r < 8; ++r) tile[half * 8 + r][mloc] = c[r];
        if (lane < 16) { sx2[lane] = x2b[jt * 16 + lane]; sdn[lane] = dnb[jt * 16 + lane]; }
        __syncthreads();
        if (lane < 16) {
            for (int col = 0; col < 16; ++col) {
                float d2 = fmaxf(x2i + sx2[col] - 2.f * tile[lane][col], 0.f) * (1.f / C2);
                float dd = sqrtf(d2);
                mx = fmaxf(mx, dd);
                if (sdn[col] > di) mn = fminf(mn, dd);
            }
        }
        __syncthreads();
    }
    if (lane < 16) { minMasked[b * N + it * 16 + lane] = mn; smax[lane] = mx; }
    __syncthreads();
    if (lane == 0) {
        float m = 0.f;
        for (int q = 0; q < 16; ++q) m = fmaxf(m, smax[q]);
        atomicMax((unsigned int*)&distmax[b], __float_as_uint(m));  // nonneg floats
    }
}

__global__ void k_score(const float* __restrict__ minMasked, const float* __restrict__ density,
                        const float* __restrict__ distmax, float* score) {
    int i = blockIdx.x * blockDim.x + threadIdx.x;
    if (i >= B * N) return;
    int b = i / N;
    float dm = distmax[b];
    score[i] = fminf(minMasked[i], dm) * density[i];
}

// ---------------------------------------------------------------------------
// top-NS selection (iterative argmax per batch, jax.top_k tie rule: lower idx)
// ---------------------------------------------------------------------------
__global__ void k_topk(const float* __restrict__ score, int* index_down) {
    int b = blockIdx.x, t = threadIdx.x;
    __shared__ float s[N];
    __shared__ float rv[256];
    __shared__ int   ri[256];
    for (int j = t; j < N; j += 256) s[j] = score[b * N + j];
    __syncthreads();
    for (int it = 0; it < NS; ++it) {
        float v = -FINF; int idx = 0;
        for (int j = t; j < N; j += 256)
            if (s[j] > v) { v = s[j]; idx = j; }
        rv[t] = v; ri[t] = idx; __syncthreads();
        for (int off = 128; off; off >>= 1) {
            if (t < off) {
                float v2 = rv[t + off]; int i2 = ri[t + off];
                if (v2 > rv[t] || (v2 == rv[t] && i2 < ri[t])) { rv[t] = v2; ri[t] = i2; }
            }
            __syncthreads();
        }
        if (t == 0) { index_down[b * NS + it] = ri[0]; s[ri[0]] = -FINF; }
        __syncthreads();
    }
}

// ---------------------------------------------------------------------------
// cluster assignment: argmin over centers (WMMA pass, tokens x centers)
// ---------------------------------------------------------------------------
__global__ void k_assign(const bf16_t* __restrict__ xtb, const float* __restrict__ x2,
                         const int* __restrict__ index_down, int* idx_cluster) {
    int b = blockIdx.y, it = blockIdx.x, lane = threadIdx.x;
    int half = lane >> 4, mloc = lane & 15, kbase = half * 8;
    const bf16_t* base = xtb + (size_t)b * N * C2;
    const float*  x2b  = x2 + b * N;
    const bf16_t* arow = base + (size_t)(it * 16 + mloc) * C2;
    v16bf a[4];
#pragma unroll
    for (int kk = 0; kk < 4; ++kk) a[kk] = load_frag(arow, kk * 32 + kbase);

    __shared__ float tile[16][16];
    __shared__ float sx2c[16];
    float x2i = (lane < 16) ? x2b[it * 16 + lane] : 0.f;
    float bestD = FINF; int bestC = 0;

    for (int jt = 0; jt < CT; ++jt) {
        int rowB = index_down[b * NS + jt * 16 + mloc];
        const bf16_t* brow = base + (size_t)rowB * C2;
        if (jt + 1 < CT)
            __builtin_prefetch(base + (size_t)index_down[b * NS + (jt + 1) * 16 + mloc] * C2, 0, 1);
        v8f c = {};
#pragma unroll
        for (int kk = 0; kk < 4; ++kk) {
            v16bf bb = load_frag(brow, kk * 32 + kbase);
            c = wmma_bf16(a[kk], bb, c);
        }
#pragma unroll
        for (int r = 0; r < 8; ++r) tile[half * 8 + r][mloc] = c[r];
        if (lane < 16) sx2c[lane] = x2b[rowB];
        __syncthreads();
        if (lane < 16) {
            for (int col = 0; col < 16; ++col) {
                float d2 = x2i + sx2c[col] - 2.f * tile[lane][col];
                if (d2 < bestD) { bestD = d2; bestC = jt * 16 + col; }
            }
        }
        __syncthreads();
    }
    if (lane < 16) idx_cluster[b * N + it * 16 + lane] = bestC;
}

__global__ void k_centerfix(const int* __restrict__ index_down, int* idx_cluster) {
    int j = blockIdx.x * blockDim.x + threadIdx.x;
    if (j >= B * NS) return;
    int b = j / NS, c = j % NS;
    idx_cluster[b * N + index_down[j]] = c;
}

// ---------------------------------------------------------------------------
// merge tokens
// ---------------------------------------------------------------------------
__global__ void k_merge_w(const int* __restrict__ idx_cluster, const float* __restrict__ weight,
                          float* allw_cl) {
    int i = blockIdx.x * blockDim.x + threadIdx.x;
    if (i >= B * N) return;
    int b = i / N;
    atomicAdd(&allw_cl[b * NS + idx_cluster[i]], weight[i]);
}

__global__ void k_merge_x(const float* __restrict__ xt, const int* __restrict__ idx_cluster,
                          const float* __restrict__ weight, const float* __restrict__ allw_cl,
                          float* x_down, float* normw) {
    int i = blockIdx.x;            // token
    int c = threadIdx.x;           // 0..127
    int b = i / N;
    int cl = idx_cluster[i];
    float nw = weight[i] / (allw_cl[b * NS + cl] + 1e-6f);
    atomicAdd(&x_down[((size_t)(b * NS + cl)) * C2 + c], xt[(size_t)i * C2 + c] * nw);
    if (c == 0) normw[i] = nw;
}

__global__ void k_final1(const int* __restrict__ idxagg, const int* __restrict__ idx_cluster,
                         const float* __restrict__ normw, const float* __restrict__ aggw,
                         float* out_idx, float* awd, float* awMax) {
    int i = blockIdx.x * blockDim.x + threadIdx.x;
    if (i >= B * N) return;
    int b = i / N;
    int tok = idxagg[i];
    out_idx[i] = (float)idx_cluster[b * N + tok];
    float v = aggw[i] * normw[b * N + tok];
    awd[i] = v;
    atomicMax((unsigned int*)&awMax[b], __float_as_uint(v));   // v > 0
}

__global__ void k_final2(const float* __restrict__ awd, const float* __restrict__ awMax,
                         float* out_aw) {
    int i = blockIdx.x * blockDim.x + threadIdx.x;
    if (i >= B * N) return;
    out_aw[i] = awd[i] / awMax[i / N];
}

// ---------------------------------------------------------------------------
// launch
// ---------------------------------------------------------------------------
extern "C" void kernel_launch(void* const* d_in, const int* in_sizes, int n_in,
                              void* d_out, int out_size, void* d_ws, size_t ws_size,
                              hipStream_t stream) {
    const float* x      = (const float*)d_in[0];
    const float* loc    = (const float*)d_in[1];
    const int*   idxagg = (const int*)d_in[2];
    const float* aggw   = (const float*)d_in[3];
    const float* convw  = (const float*)d_in[4];
    const float* convb  = (const float*)d_in[5];
    const float* skipw  = (const float*)d_in[6];
    const float* lng    = (const float*)d_in[7];
    const float* lnb    = (const float*)d_in[8];
    const float* confw  = (const float*)d_in[9];
    const float* confb  = (const float*)d_in[10];
    float* out = (float*)d_out;

    // workspace layout (float slots; every region is a multiple of 4 -> 16B aligned)
    float* ws = (float*)d_ws;
    size_t off = 0;
    auto take = [&](size_t n) { float* p = ws + off; off += n; return p; };
    float* allw_grid = take((size_t)B * H * W);         // 12544
    float* xmap      = take((size_t)B * H * W * C);     // 802816
    float* conv_out  = take((size_t)B * P2 * C2);       // 401408
    float* allw_tok  = take((size_t)B * N);             // 12544
    float* xt        = take((size_t)B * N * C2);        // 1605632
    float* x2        = take((size_t)B * N);
    float* density   = take((size_t)B * N);
    float* minMasked = take((size_t)B * N);
    float* score     = take((size_t)B * N);
    float* weight    = take((size_t)B * N);
    float* normw     = take((size_t)B * N);
    float* allw_cl   = take((size_t)B * NS);            // 3136
    float* distmax   = take(4);
    float* awMax     = take(4);
    float* awd       = take((size_t)B * N);
    int*   idx_cl    = (int*)take((size_t)B * N);
    int*   index_dn  = (int*)take((size_t)B * NS);
    bf16_t* x_bf     = (bf16_t*)take((size_t)B * N * C / 2);      // bf16, half slots
    bf16_t* w_bf     = (bf16_t*)take((size_t)C2 * C / 2);
    bf16_t* xt_bf    = (bf16_t*)take((size_t)B * N * C2 / 2);
    size_t total_ws_f = off;

    // d_out layout: [x_down | idx_agg_down(as float) | agg_weight_down]
    float* out_x   = out;
    float* out_idx = out + (size_t)B * NS * C2;
    float* out_aw  = out_idx + (size_t)B * N;

    // 0) zero workspace + accumulated output region
    k_fill<<<4096, 256, 0, stream>>>(ws, 0.f, (int)total_ws_f);
    k_fill<<<1024, 256, 0, stream>>>(out_x, 0.f, B * NS * C2);

    // 1) token2map
    k_t2m_w<<<(B * N + 255) / 256, 256, 0, stream>>>(loc, allw_grid);
    k_t2m_s<<<B * N, C, 0, stream>>>(x, loc, idxagg, allw_grid, xmap);

    // 2) conv 3x3 s2 (64 -> 128)
    k_conv<<<B * P2, C2, 0, stream>>>(xmap, convw, convb, conv_out);

    // 3) map2token into xt
    k_m2t_w<<<(B * N + 255) / 256, 256, 0, stream>>>(idxagg, aggw, allw_tok);
    k_m2t_s<<<B * N, C2, 0, stream>>>(conv_out, loc, idxagg, aggw, allw_tok, xt);

    // 4) skip projection xt += x @ skip_w.T  (WMMA bf16)
    k_cvt<<<1024, 256, 0, stream>>>(x, x_bf, B * N * C);
    k_cvt<<<32, 256, 0, stream>>>(skipw, w_bf, C2 * C);
    k_skip<<<dim3((B * N) / 16, C2 / 16), 32, 0, stream>>>(x_bf, w_bf, xt);

    // 5) LayerNorm + conf/weight + bf16 features + row norms
    k_ln<<<B * N, C2, 0, stream>>>(xt, xt_bf, lng, lnb, confw, confb, weight, x2);

    // 6) DPC-KNN: density (WMMA Gram pass 1, async LDS double-buffered)
    k_dens<<<dim3(NT, B), 32, 0, stream>>>(xt_bf, x2, density);

    // 7) masked min dist + dist_max (WMMA Gram pass 2), then score
    k_dind<<<dim3(NT, B), 32, 0, stream>>>(xt_bf, x2, density, minMasked, distmax);
    k_score<<<(B * N + 255) / 256, 256, 0, stream>>>(minMasked, density, distmax, score);

    // 8) top-NS cluster centers
    k_topk<<<B, 256, 0, stream>>>(score, index_dn);

    // 9) assign tokens to nearest center (WMMA pass 3) + centers own cluster
    k_assign<<<dim3(NT, B), 32, 0, stream>>>(xt_bf, x2, index_dn, idx_cl);
    k_centerfix<<<(B * NS + 255) / 256, 256, 0, stream>>>(index_dn, idx_cl);

    // 10) merge tokens -> x_down, and final outputs
    k_merge_w<<<(B * N + 255) / 256, 256, 0, stream>>>(idx_cl, weight, allw_cl);
    k_merge_x<<<B * N, C2, 0, stream>>>(xt, idx_cl, weight, allw_cl, out_x, normw);
    k_final1<<<(B * N + 255) / 256, 256, 0, stream>>>(idxagg, idx_cl, normw, aggw,
                                                      out_idx, awd, awMax);
    k_final2<<<(B * N + 255) / 256, 256, 0, stream>>>(awd, awMax, out_aw);
}
